// RWKV_GPT_49031346651565
// MI455X (gfx1250) — matrix operand order repaired and checked by reading, not compiled, not run on hardware
//
#include <hip/hip_runtime.h>
#include <hip/hip_bf16.h>

// ---------------- problem constants (match reference) ----------------
#define BSZ   2
#define TLEN  2048
#define CDIM  1024
#define HDIM  4096
#define VDIM  32000
#define LNUM  6
#define MROWS (BSZ * TLEN)   // 4096 token rows
#define LN_EPSF 1e-5f

typedef __attribute__((ext_vector_type(16))) __bf16 v16bf;
typedef __attribute__((ext_vector_type(8)))  __bf16 v8bf;
typedef __attribute__((ext_vector_type(8)))  float  v8f;

// async-to-LDS builtin operand types: 16-byte int vector in AS(1)/AS(3)
typedef __attribute__((vector_size(16))) int v4i32;
typedef __attribute__((address_space(1))) v4i32 v4i32_g;
typedef __attribute__((address_space(3))) v4i32 v4i32_l;

#if __has_builtin(__builtin_amdgcn_global_load_async_to_lds_b128)
#define HAVE_ASYNC 1
#else
#define HAVE_ASYNC 0
#endif

__device__ inline void wait_async_all() {
#if HAVE_ASYNC
#if __has_builtin(__builtin_amdgcn_s_wait_asynccnt)
    __builtin_amdgcn_s_wait_asynccnt(0);
#else
    asm volatile("s_wait_asynccnt 0x0" ::: "memory");
#endif
#endif
}

// ---------------------------------------------------------------------
// f32 -> bf16 convert (used for weights and the head input)
__global__ __launch_bounds__(256) void k_f32_to_bf16(const float* __restrict__ in,
                                                     __bf16* __restrict__ out, long n) {
    long i = (long)blockIdx.x * blockDim.x + threadIdx.x;
    if (i < n) out[i] = (__bf16)in[i];
}

// embedding gather: x[m,:] = emb[idx[m],:]
__global__ __launch_bounds__(256) void k_gather(const int* __restrict__ idx,
                                                const float* __restrict__ emb,
                                                float* __restrict__ x) {
    int m = blockIdx.x;
    const float* src = emb + (size_t)idx[m] * CDIM;
    float* dst = x + (size_t)m * CDIM;
    for (int c = threadIdx.x; c < CDIM; c += blockDim.x) dst[c] = src[c];
}

// row LayerNorm: y[m,:] = (x[m,:]-mean)*rsqrt(var+eps)*w + b   (in-place safe)
__global__ __launch_bounds__(256) void k_layernorm(const float* __restrict__ x,
                                                   float* __restrict__ y,
                                                   const float* __restrict__ w,
                                                   const float* __restrict__ b) {
    __shared__ float s1[256], s2[256];
    int m = blockIdx.x;
    const float* row = x + (size_t)m * CDIM;
    float vals[CDIM / 256];
    float sum = 0.f, sq = 0.f;
#pragma unroll
    for (int i = 0; i < CDIM / 256; i++) {
        float v = row[threadIdx.x + i * 256];
        vals[i] = v; sum += v; sq += v * v;
    }
    s1[threadIdx.x] = sum; s2[threadIdx.x] = sq; __syncthreads();
    for (int off = 128; off > 0; off >>= 1) {
        if (threadIdx.x < off) { s1[threadIdx.x] += s1[threadIdx.x + off];
                                 s2[threadIdx.x] += s2[threadIdx.x + off]; }
        __syncthreads();
    }
    float mean = s1[0] * (1.f / CDIM);
    float var  = s2[0] * (1.f / CDIM) - mean * mean;
    float inv  = rsqrtf(var + LN_EPSF);
    float* out = y + (size_t)m * CDIM;
#pragma unroll
    for (int i = 0; i < CDIM / 256; i++) {
        int c = threadIdx.x + i * 256;
        out[c] = (vals[i] - mean) * inv * w[c] + b[c];
    }
}

// time-shift mix: out = h*mu + h_prev*(1-mu), h_prev = 0 at t==0; emits bf16 GEMM operands.
__global__ __launch_bounds__(256) void k_mix(const float* __restrict__ h,
                                             const float* __restrict__ mk,
                                             const float* __restrict__ mv,
                                             const float* __restrict__ mr,
                                             __bf16* __restrict__ ok,
                                             __bf16* __restrict__ ov,
                                             __bf16* __restrict__ orr) {
    long i = (long)blockIdx.x * blockDim.x + threadIdx.x;
    if (i >= (long)MROWS * CDIM) return;
    int  c = (int)(i % CDIM);
    long m = i / CDIM;
    int  t = (int)(m % TLEN);
    float hv = h[i];
    float hp = (t == 0) ? 0.f : h[i - CDIM];
    float a = mk[c];
    ok[i] = (__bf16)(hv * a + hp * (1.f - a));
    if (ov) { float bm = mv[c]; ov[i] = (__bf16)(hv * bm + hp * (1.f - bm)); }
    float rm = mr[c];
    orr[i] = (__bf16)(hv * rm + hp * (1.f - rm));
}

// ---------------------------------------------------------------------
// WMMA bf16 GEMM:  D[M,N] (f32) = A[M,K] (bf16, row-major) x W[N,K]^T (bf16, row-major)
// Block = 8 waves, block tile 128x128. B (weight) tile staged in LDS via async-to-LDS,
// double buffered; each wave computes 16 rows x 128 cols (8x v_wmma_f32_16x16x32_bf16/K-step).

#define BROWSTRIDE 40   // __bf16 elems per LDS B-tile row (80 B, padded: banks 20r%64 distinct)
#define BTILE_ELEMS (128 * BROWSTRIDE)

// A fragment (16x32, 16-bit A layout per ISA 7.12.2):
//   lanes 0-15 : row M=lane,    K {k0+0..7, k0+16..23}
//   lanes 16-31: row M=lane-16, K {k0+8..15, k0+24..31}
__device__ inline v16bf load_fragA(const __bf16* __restrict__ base, int row, int k0,
                                   int ld, int half) {
    const __bf16* p = base + (size_t)row * ld + k0 + half * 8;
    v8bf lo = *(const v8bf*)(p);
    v8bf hi = *(const v8bf*)(p + 16);
    return __builtin_shufflevector(lo, hi, 0,1,2,3,4,5,6,7,8,9,10,11,12,13,14,15);
}

// B fragment (32x16) from LDS tile: column N = j*16 + (lane&15);
// lanes 0-15 hold K {k0+0..7, k0+16..23}, lanes 16-31 hold K {k0+8..15, k0+24..31}.
__device__ inline v16bf lds_fragB(const __bf16* buf, int j, int r, int half) {
    const __bf16* p = buf + (j * 16 + r) * BROWSTRIDE + half * 8;
    v8bf lo = *(const v8bf*)(p);
    v8bf hi = *(const v8bf*)(p + 16);
    return __builtin_shufflevector(lo, hi, 0,1,2,3,4,5,6,7,8,9,10,11,12,13,14,15);
}

// Stage B tile: W rows n0..n0+127, K cols k0..k0+31 (64 B/row) -> LDS (80 B row stride).
// 512 x 16-byte chunks, 2 per thread.
__device__ inline void issue_tileB(const __bf16* __restrict__ W, int n0, int k0, int Kv,
                                   __bf16* buf, int tid) {
#pragma unroll
    for (int q = 0; q < 2; q++) {
        int chunk = tid * 2 + q;          // 0..511
        int row = chunk >> 2;             // 0..127
        int c16 = chunk & 3;              // 16-byte chunk within 64-byte row
        const __bf16* g = W + (size_t)(n0 + row) * Kv + k0 + c16 * 8;
        __bf16* l = buf + row * BROWSTRIDE + c16 * 8;
#if HAVE_ASYNC
        __builtin_amdgcn_global_load_async_to_lds_b128((v4i32_g*)g, (v4i32_l*)l, 0, 0);
#else
        *(uint4*)l = *(const uint4*)g;    // sync fallback: global load + ds store
#endif
    }
}

__global__ __launch_bounds__(256) void k_gemm_bf16(const __bf16* __restrict__ A,
                                                   const __bf16* __restrict__ W,
                                                   float* __restrict__ D,
                                                   int Mv, int Nv, int Kv) {
    __shared__ __bf16 ldsB[2][BTILE_ELEMS];
    const int tid  = threadIdx.x;
    const int wave = tid >> 5;
    const int ln   = tid & 31;
    const int half = ln >> 4;
    const int r    = ln & 15;
    const int Nb   = Nv >> 7;
    const int m0   = (blockIdx.x / Nb) << 7;
    const int n0   = (blockIdx.x % Nb) << 7;
    const int mw   = m0 + wave * 16;         // this wave's 16 output rows

    v8f c[8];
#pragma unroll
    for (int j = 0; j < 8; j++) c[j] = (v8f){};

    issue_tileB(W, n0, 0, Kv, &ldsB[0][0], tid);
    wait_async_all();
    __syncthreads();

    const int steps = Kv >> 5;
    for (int s = 0; s < steps; s++) {
        const int cur = s & 1;
        if (s + 1 < steps)
            issue_tileB(W, n0, (s + 1) << 5, Kv, &ldsB[cur ^ 1][0], tid);

        v16bf a = load_fragA(A, mw + r, s << 5, Kv, half);
#pragma unroll
        for (int j = 0; j < 8; j++) {
            v16bf b = lds_fragB(&ldsB[cur][0], j, r, half);
            c[j] = __builtin_amdgcn_wmma_f32_16x16x32_bf16(false, a, false, b,
                                                           (short)0, c[j], false, false);
        }
        wait_async_all();
        __syncthreads();
    }

    // C/D layout: VGPR v, lanes 0-15 -> row mw+v, lanes 16-31 -> row mw+v+8; col = n0+j*16+(lane&15)
#pragma unroll
    for (int j = 0; j < 8; j++) {
#pragma unroll
        for (int v = 0; v < 8; v++) {
            int row = mw + v + (half << 3);
            D[(size_t)row * Nv + n0 + j * 16 + r] = c[j][v];
        }
    }
}

// ---------------------------------------------------------------------
// WKV recurrence: one thread per (batch, channel); sequential over T.
__global__ __launch_bounds__(256) void k_wkv(const float* __restrict__ kbuf,
                                             const float* __restrict__ vbuf,
                                             const float* __restrict__ td,
                                             const float* __restrict__ tf,
                                             float* __restrict__ wkv) {
    int i = blockIdx.x * blockDim.x + threadIdx.x;
    if (i >= BSZ * CDIM) return;
    int b = i / CDIM, c = i % CDIM;
    float w = -__expf(td[c]);
    float u = tf[c];
    float aa = 0.f, bb = 0.f, pp = -1e38f;
    const float* kp = kbuf + (size_t)b * TLEN * CDIM + c;
    const float* vp = vbuf + (size_t)b * TLEN * CDIM + c;
    float*       yp = wkv  + (size_t)b * TLEN * CDIM + c;
    for (int t = 0; t < TLEN; t++) {
        float kt = kp[(size_t)t * CDIM];
        float vt = vp[(size_t)t * CDIM];
        float ww = u + kt;
        float q  = fmaxf(pp, ww);
        float e1 = __expf(pp - q);
        float e2 = __expf(ww - q);
        yp[(size_t)t * CDIM] = (e1 * aa + e2 * vt) / (e1 * bb + e2);
        float ww2 = pp + w;
        float q2  = fmaxf(ww2, kt);
        float e1b = __expf(ww2 - q2);
        float e2b = __expf(kt - q2);
        aa = e1b * aa + e2b * vt;
        bb = e1b * bb + e2b;
        pp = q2;
    }
}

// sigmoid(r)*wkv -> bf16 operand for Wo GEMM
__global__ __launch_bounds__(256) void k_att_in(const float* __restrict__ r,
                                                const float* __restrict__ wkv,
                                                __bf16* __restrict__ out, long n) {
    long i = (long)blockIdx.x * blockDim.x + threadIdx.x;
    if (i < n) {
        float s = 1.f / (1.f + __expf(-r[i]));
        out[i] = (__bf16)(s * wkv[i]);
    }
}

__global__ __launch_bounds__(256) void k_add(float* __restrict__ x,
                                             const float* __restrict__ y, long n) {
    long i = (long)blockIdx.x * blockDim.x + threadIdx.x;
    if (i < n) x[i] += y[i];
}

// relu(k)^2 -> bf16 operand for Wv GEMM
__global__ __launch_bounds__(256) void k_relusq(const float* __restrict__ in,
                                                __bf16* __restrict__ out, long n) {
    long i = (long)blockIdx.x * blockDim.x + threadIdx.x;
    if (i < n) {
        float v = fmaxf(in[i], 0.f);
        out[i] = (__bf16)(v * v);
    }
}

// x += sigmoid(r2) * kv
__global__ __launch_bounds__(256) void k_ffn_out(float* __restrict__ x,
                                                 const float* __restrict__ r,
                                                 const float* __restrict__ kv, long n) {
    long i = (long)blockIdx.x * blockDim.x + threadIdx.x;
    if (i < n) {
        float s = 1.f / (1.f + __expf(-r[i]));
        x[i] += s * kv[i];
    }
}

// ---------------------------------------------------------------------
extern "C" void kernel_launch(void* const* d_in, const int* in_sizes, int n_in,
                              void* d_out, int out_size, void* d_ws, size_t ws_size,
                              hipStream_t stream) {
    (void)in_sizes; (void)n_in; (void)out_size; (void)ws_size;

    const int*   idx   = (const int*)  d_in[0];
    const float* emb   = (const float*)d_in[1];
    const float* ln0w  = (const float*)d_in[2],  *ln0b = (const float*)d_in[3];
    const float* ln1w  = (const float*)d_in[4],  *ln1b = (const float*)d_in[5];
    const float* ln2w  = (const float*)d_in[6],  *ln2b = (const float*)d_in[7];
    const float* td    = (const float*)d_in[8],  *tf   = (const float*)d_in[9];
    const float* tmk   = (const float*)d_in[10], *tmv  = (const float*)d_in[11], *tmr = (const float*)d_in[12];
    const float* attWk = (const float*)d_in[13], *attWv = (const float*)d_in[14];
    const float* attWr = (const float*)d_in[15], *attWo = (const float*)d_in[16];
    const float* fmk   = (const float*)d_in[17], *fmr  = (const float*)d_in[18];
    const float* ffnWk = (const float*)d_in[19], *ffnWr = (const float*)d_in[20], *ffnWv = (const float*)d_in[21];
    const float* lnow  = (const float*)d_in[22], *lnob = (const float*)d_in[23];
    const float* headw = (const float*)d_in[24];
    float* logits = (float*)d_out;                    // [M, V] == [B,T,V]

    // ---- workspace carve-out ----
    char* ws = (char*)d_ws;
    size_t off = 0;
    auto alloc = [&](size_t bytes) -> char* {
        char* p = ws + off;
        off = (off + bytes + 255) & ~(size_t)255;
        return p;
    };
    const size_t CC = (size_t)CDIM * CDIM;      // 1M
    const size_t HC = (size_t)HDIM * CDIM;      // 4M
    const size_t MC = (size_t)MROWS * CDIM;     // 4M
    const size_t MH = (size_t)MROWS * HDIM;     // 16M

    __bf16* wbAttWk = (__bf16*)alloc(LNUM * CC * 2);
    __bf16* wbAttWv = (__bf16*)alloc(LNUM * CC * 2);
    __bf16* wbAttWr = (__bf16*)alloc(LNUM * CC * 2);
    __bf16* wbAttWo = (__bf16*)alloc(LNUM * CC * 2);
    __bf16* wbFfnWk = (__bf16*)alloc(LNUM * HC * 2);
    __bf16* wbFfnWr = (__bf16*)alloc(LNUM * CC * 2);
    __bf16* wbFfnWv = (__bf16*)alloc(LNUM * HC * 2);
    __bf16* wbHead  = (__bf16*)alloc((size_t)VDIM * CDIM * 2);

    float*  x    = (float*) alloc(MC * 4);
    float*  h    = (float*) alloc(MC * 4);
    __bf16* xkB  = (__bf16*)alloc(MC * 2);
    __bf16* xvB  = (__bf16*)alloc(MC * 2);   // reused as attin operand
    __bf16* xrB  = (__bf16*)alloc(MC * 2);
    float*  kbuf = (float*) alloc(MC * 4);   // reused as attout
    float*  vbuf = (float*) alloc(MC * 4);   // reused as kv
    float*  rbuf = (float*) alloc(MC * 4);   // reused as r2
    float*  wkvb = (float*) alloc(MC * 4);
    float*  kf   = (float*) alloc(MH * 4);   // ffn pre-activation
    __bf16* kkB  = (__bf16*)alloc(MH * 2);   // relu^2 operand
    __bf16* hB   = (__bf16*)alloc(MC * 2);   // head input

    auto grid1 = [](long n) { return dim3((unsigned)((n + 255) / 256)); };
    auto conv  = [&](const float* src, __bf16* dst, long n) {
        k_f32_to_bf16<<<grid1(n), 256, 0, stream>>>(src, dst, n);
    };
    auto gemm = [&](const __bf16* A, const __bf16* W, float* D, int Mv, int Nv, int Kv) {
        int blocks = (Mv / 128) * (Nv / 128);
        k_gemm_bf16<<<blocks, 256, 0, stream>>>(A, W, D, Mv, Nv, Kv);
    };

    // ---- weights -> bf16 (every call; deterministic) ----
    conv(attWk, wbAttWk, (long)LNUM * CC);
    conv(attWv, wbAttWv, (long)LNUM * CC);
    conv(attWr, wbAttWr, (long)LNUM * CC);
    conv(attWo, wbAttWo, (long)LNUM * CC);
    conv(ffnWk, wbFfnWk, (long)LNUM * HC);
    conv(ffnWr, wbFfnWr, (long)LNUM * CC);
    conv(ffnWv, wbFfnWv, (long)LNUM * HC);
    conv(headw, wbHead,  (long)VDIM * CDIM);

    // ---- embed + ln0 ----
    k_gather<<<MROWS, 256, 0, stream>>>(idx, emb, x);
    k_layernorm<<<MROWS, 256, 0, stream>>>(x, x, ln0w, ln0b);

    for (int i = 0; i < LNUM; i++) {
        // ---- TimeMix ----
        k_layernorm<<<MROWS, 256, 0, stream>>>(x, h, ln1w + i * CDIM, ln1b + i * CDIM);
        k_mix<<<grid1((long)MC), 256, 0, stream>>>(h, tmk + i * CDIM, tmv + i * CDIM,
                                                   tmr + i * CDIM, xkB, xvB, xrB);
        gemm(xkB, wbAttWk + (size_t)i * CC, kbuf, MROWS, CDIM, CDIM);
        gemm(xvB, wbAttWv + (size_t)i * CC, vbuf, MROWS, CDIM, CDIM);
        gemm(xrB, wbAttWr + (size_t)i * CC, rbuf, MROWS, CDIM, CDIM);
        k_wkv<<<(BSZ * CDIM + 255) / 256, 256, 0, stream>>>(kbuf, vbuf,
                                                            td + i * CDIM, tf + i * CDIM, wkvb);
        k_att_in<<<grid1((long)MC), 256, 0, stream>>>(rbuf, wkvb, xvB, (long)MC);
        gemm(xvB, wbAttWo + (size_t)i * CC, kbuf, MROWS, CDIM, CDIM);
        k_add<<<grid1((long)MC), 256, 0, stream>>>(x, kbuf, (long)MC);

        // ---- ChannelMix ----
        k_layernorm<<<MROWS, 256, 0, stream>>>(x, h, ln2w + i * CDIM, ln2b + i * CDIM);
        k_mix<<<grid1((long)MC), 256, 0, stream>>>(h, fmk + i * CDIM, nullptr,
                                                   fmr + i * CDIM, xkB, nullptr, xrB);
        gemm(xkB, wbFfnWk + (size_t)i * HC, kf, MROWS, HDIM, CDIM);
        k_relusq<<<grid1((long)MH), 256, 0, stream>>>(kf, kkB, (long)MH);
        gemm(xrB, wbFfnWr + (size_t)i * CC, rbuf, MROWS, CDIM, CDIM);
        gemm(kkB, wbFfnWv + (size_t)i * HC, vbuf, MROWS, CDIM, HDIM);
        k_ffn_out<<<grid1((long)MC), 256, 0, stream>>>(x, rbuf, vbuf, (long)MC);
    }

    // ---- head ----
    k_layernorm<<<MROWS, 256, 0, stream>>>(x, h, lnow, lnob);
    k_f32_to_bf16<<<grid1((long)MC), 256, 0, stream>>>(h, hB, (long)MC);
    gemm(hB, wbHead, logits, MROWS, VDIM, CDIM);
}
